// ScoringLayer_36120674959466
// MI455X (gfx1250) — matrix operand order repaired and checked by reading, not compiled, not run on hardware
//
#include <hip/hip_runtime.h>
#include <hip/hip_bf16.h>
#include <math.h>

// Problem constants (fixed by setup_inputs)
#define NB 32
#define NW 512
#define NL 16
#define ND 768

#define K1 0.1f
#define B_PARAM 1.2f
#define AVDL 50.0f
#define EPS 1e-8f

typedef __attribute__((ext_vector_type(16))) __bf16 v16bf;
typedef __attribute__((ext_vector_type(8)))  float  v8f;
typedef __attribute__((ext_vector_type(2)))  float  f2;

// ---------------------------------------------------------------------------
// Kernel 1: per-window 64-bit FNV hash + null mask.
//   q side -> hq, qmask ; d side -> hd, tfm = d_mask * tf
// ---------------------------------------------------------------------------
__global__ __launch_bounds__(256) void hash_mask_kernel(
    const int* __restrict__ qids, const int* __restrict__ dids,
    const int* __restrict__ dtfs,
    unsigned long long* __restrict__ hq, unsigned long long* __restrict__ hd,
    float* __restrict__ qmask, float* __restrict__ tfm) {
  int idx = blockIdx.x * blockDim.x + threadIdx.x;  // 0 .. 2*NB*NW-1
  if (idx >= 2 * NB * NW) return;
  bool isQ = idx < NB * NW;
  int w = isQ ? idx : idx - NB * NW;
  const int* p = (isQ ? qids : dids) + (size_t)w * NL;
  unsigned long long h = 1469598103934665603ULL;
  int nz = 0;
#pragma unroll
  for (int i = 0; i < NL; ++i) {
    int id = p[i];
    nz |= id;
    h = (h ^ (unsigned long long)(unsigned)id) * 1099511628211ULL;
  }
  float m = nz ? 1.0f : 0.0f;
  if (isQ) {
    hq[w] = h;
    qmask[w] = m;
  } else {
    hd[w] = h;
    tfm[w] = m * (float)dtfs[w];
  }
}

// ---------------------------------------------------------------------------
// Kernel 2: v[b,d] = sum_k tfm[b,k] * d_rep[b,k,d]   (streams all of d_rep)
// grid: NB*3 blocks x 256 threads, each thread owns one column d.
// ---------------------------------------------------------------------------
__global__ __launch_bounds__(256) void vmat_kernel(
    const float* __restrict__ drep, const float* __restrict__ tfm,
    float* __restrict__ vout) {
  int blk = blockIdx.x;
  int b = blk / 3;
  int d = (blk % 3) * 256 + threadIdx.x;
  const float* base = drep + (size_t)b * NW * ND + d;
  const float* tfb = tfm + b * NW;
  float acc = 0.0f;
#pragma unroll 4
  for (int k = 0; k < NW; ++k) {
    acc = fmaf(tfb[k], base[(size_t)k * ND], acc);
  }
  vout[b * ND + d] = acc;
}

// ---------------------------------------------------------------------------
// Kernel 3: dot_ws[b,q] = qmask * (q_rep[b,q,:] . v[b,:]) via WMMA bf16.
// Split-bf16 (hi/lo) x3 WMMAs per K-chunk for ~fp32 accuracy.
// One wave per 16-row tile; B = v broadcast into all 16 columns.
// ---------------------------------------------------------------------------
__global__ __launch_bounds__(256) void qdot_wmma_kernel(
    const float* __restrict__ qrep, const float* __restrict__ vmat,
    const float* __restrict__ qmask, float* __restrict__ dotws) {
  int wave = threadIdx.x >> 5;
  int tile = blockIdx.x * 8 + wave;  // 0 .. 1023
  int b = tile >> 5;                 // 32 tiles of 16 rows per batch
  int row0 = (tile & 31) * 16;
  int lane = threadIdx.x & 31;
  int r = lane & 15;
  int hi = lane >> 4;

  const float* Arow = qrep + ((size_t)b * NW + row0 + r) * ND;
  const float* vb = vmat + b * ND;

  v8f c = {};
  for (int k0 = 0; k0 < ND; k0 += 32) {
    // A fragment (16x32 bf16): VGPR j<4 -> K=2j(+1)+8*hi ; j>=4 -> +16
    v16bf ah, al;
#pragma unroll
    for (int j = 0; j < 8; ++j) {
      int kk = k0 + ((j >= 4) ? 16 : 0) + 2 * (j & 3) + hi * 8;
      f2 x = *(const f2*)(Arow + kk);
      __bf16 h0 = (__bf16)x.x, h1 = (__bf16)x.y;
      ah[2 * j] = h0;
      ah[2 * j + 1] = h1;
      al[2 * j] = (__bf16)(x.x - (float)h0);
      al[2 * j + 1] = (__bf16)(x.y - (float)h1);
    }
    // B fragment (32x16 bf16): same K order as A (VGPR j<4 -> K=2j(+1)+8*hi ; j>=4 -> +16),
    // v broadcast across all 16 N columns.
    v16bf bh, bl;
#pragma unroll
    for (int j = 0; j < 8; ++j) {
      int kk = k0 + ((j >= 4) ? 16 : 0) + 2 * (j & 3) + hi * 8;
      f2 y = *(const f2*)(vb + kk);
      __bf16 h0 = (__bf16)y.x, h1 = (__bf16)y.y;
      bh[2 * j] = h0;
      bh[2 * j + 1] = h1;
      bl[2 * j] = (__bf16)(y.x - (float)h0);
      bl[2 * j + 1] = (__bf16)(y.y - (float)h1);
    }
    c = __builtin_amdgcn_wmma_f32_16x16x32_bf16(false, ah, false, bh,
                                                (short)0, c, false, false);
    c = __builtin_amdgcn_wmma_f32_16x16x32_bf16(false, ah, false, bl,
                                                (short)0, c, false, false);
    c = __builtin_amdgcn_wmma_f32_16x16x32_bf16(false, al, false, bh,
                                                (short)0, c, false, false);
  }
  // D layout: VGPR i = row (i + 8*hi), N = lane&15. All N columns identical
  // (B broadcast) -> lanes with r==0 write 8 rows each.
  if (r == 0) {
#pragma unroll
    for (int i = 0; i < 8; ++i) {
      int row = row0 + hi * 8 + i;
      dotws[b * NW + row] = c[i] * qmask[b * NW + row];
    }
  }
}

// ---------------------------------------------------------------------------
// Kernel 4: em_tf[b,q] = sum_k [window ids equal] * tf[b,k]
// Hash prefilter (exact verify on hit). One block per batch.
// ---------------------------------------------------------------------------
__global__ __launch_bounds__(256) void em_kernel(
    const int* __restrict__ qids, const int* __restrict__ dids,
    const int* __restrict__ dtfs,
    const unsigned long long* __restrict__ hq,
    const unsigned long long* __restrict__ hd, float* __restrict__ emws) {
  __shared__ unsigned long long sh[NW];
  __shared__ float stf[NW];
  int b = blockIdx.x;
  int t = threadIdx.x;
  for (int i = t; i < NW; i += 256) {
    sh[i] = hd[b * NW + i];
    stf[i] = (float)dtfs[b * NW + i];
  }
  __syncthreads();
  for (int qi = t; qi < NW; qi += 256) {
    unsigned long long h = hq[b * NW + qi];
    float acc = 0.0f;
    for (int k = 0; k < NW; ++k) {
      if (sh[k] == h) {  // essentially never taken for random ids
        const int* qp = qids + ((size_t)b * NW + qi) * NL;
        const int* dp = dids + ((size_t)b * NW + k) * NL;
        int eq = 1;
#pragma unroll
        for (int i = 0; i < NL; ++i) eq &= (qp[i] == dp[i]);
        if (eq) acc += stf[k];
      }
    }
    emws[b * NW + qi] = acc;
  }
}

// ---------------------------------------------------------------------------
// Kernel 5: BM25 transform + reduce over q windows. One block per batch.
// ---------------------------------------------------------------------------
__global__ __launch_bounds__(256) void bm25_kernel(
    const float* __restrict__ dotws, const float* __restrict__ emws,
    const int* __restrict__ dtfs, float* __restrict__ out) {
  __shared__ float red[256];
  int b = blockIdx.x, t = threadIdx.x;
  float dlp = 0.0f;
  for (int i = t; i < NW; i += 256) dlp += (float)dtfs[b * NW + i];
  red[t] = dlp;
  __syncthreads();
  for (int s = 128; s > 0; s >>= 1) {
    if (t < s) red[t] += red[t + s];
    __syncthreads();
  }
  float dl = red[0];
  __syncthreads();
  float nomc = K1 * (1.0f - B_PARAM + B_PARAM * dl / AVDL);
  float acc = 0.0f;
  for (int i = t; i < NW; i += 256) {
    float dtf = dotws[b * NW + i] + emws[b * NW + i];
    float denom = dtf + dtf * K1;
    float nom = dtf + nomc;
    acc += denom / (nom + EPS);
  }
  red[t] = acc;
  __syncthreads();
  for (int s = 128; s > 0; s >>= 1) {
    if (t < s) red[t] += red[t + s];
    __syncthreads();
  }
  if (t == 0) out[b] = red[0];
}

// ---------------------------------------------------------------------------
extern "C" void kernel_launch(void* const* d_in, const int* in_sizes, int n_in,
                              void* d_out, int out_size, void* d_ws,
                              size_t ws_size, hipStream_t stream) {
  const float* qrep = (const float*)d_in[0];
  const float* drep = (const float*)d_in[1];
  const int* qids = (const int*)d_in[2];
  const int* dids = (const int*)d_in[3];
  const int* dtfs = (const int*)d_in[4];
  float* out = (float*)d_out;

  // Workspace layout (needs ~608 KB)
  char* ws = (char*)d_ws;
  unsigned long long* hq = (unsigned long long*)(ws + 0);       // 128 KB
  unsigned long long* hd = (unsigned long long*)(ws + 131072);  // 128 KB
  float* qmask = (float*)(ws + 262144);                         // 64 KB
  float* tfm = (float*)(ws + 327680);                           // 64 KB
  float* vmat = (float*)(ws + 393216);                          // 96 KB
  float* dotws = (float*)(ws + 491520);                         // 64 KB
  float* emws = (float*)(ws + 557056);                          // 64 KB

  hash_mask_kernel<<<(2 * NB * NW + 255) / 256, 256, 0, stream>>>(
      qids, dids, dtfs, hq, hd, qmask, tfm);
  vmat_kernel<<<NB * 3, 256, 0, stream>>>(drep, tfm, vmat);
  qdot_wmma_kernel<<<(NB * NW / 16) / 8, 256, 0, stream>>>(qrep, vmat, qmask,
                                                           dotws);
  em_kernel<<<NB, 256, 0, stream>>>(qids, dids, dtfs, hq, hd, emws);
  bm25_kernel<<<NB, 256, 0, stream>>>(dotws, emws, dtfs, out);
}
